// SlimTikNetworkLinearOperatorFull_34617436406084
// MI455X (gfx1250) — matrix operand order repaired and checked by reading, not compiled, not run on hardware
//
#include <hip/hip_runtime.h>
#include <math.h>

// ---------------- problem constants (from reference) ----------------
#define INC    16
#define OUTC   16
#define KSZ    3
#define NBAT   8
#define HH     16
#define WW     16
#define HO     18
#define WO     18
#define NUMZ   (NBAT*OUTC*HO*WO)     // 41472
#define NWC    (INC*OUTC*KSZ*KSZ)    // 2304
#define NW     (NWC+OUTC)            // 2320
#define NROWB  (NBAT*HO*WO)          // 2592
#define SDIM   (INC*KSZ*KSZ)         // 144
#define NLAM   11
#define NPAIRS (NW/2)                // 1160
#define NSWEEPS 4
#define SUM_LAMBDA_C 0.05f

typedef __attribute__((ext_vector_type(16))) __bf16 v16bf;
typedef __attribute__((ext_vector_type(8)))  float  v8f;
typedef __attribute__((ext_vector_type(8)))  int    v8i;
typedef __attribute__((ext_vector_type(4)))  int    v4i;

__device__ __forceinline__ unsigned short f2bf(float f) {
  unsigned int u = __float_as_uint(f);
  u += 0x7FFFu + ((u >> 16) & 1u);          // round-to-nearest-even
  return (unsigned short)(u >> 16);
}

__device__ __forceinline__ float lamval(int l) {
  float v = 0.05f * powf(10.f, -1.f + 0.2f * (float)l);
  return fminf(fmaxf(v, 1e-7f), 1000.f);
}

// ---------------- generic bf16 WMMA GEMM: C = alpha*A^T B + betac*C -------
// A_elem(k,m) = A[k*rsA + m*csA]   (bf16 raw bits)
// B_elem(k,n) = B[k*rsB + n*csB]
// Block tile 128x64, K-step 32. 256 threads = 8 waves in a 4x2 grid; each
// wave owns a 32x32 tile: 2 A-frags x 2 B-frags -> 4 v_wmma per K-step.
// LDS tiles are K-pair-packed, M/N-major: Apk[m][kp] = bf16(2kp)|bf16(2kp+1)<<16,
// so every WMMA fragment is exactly two 16B-aligned ds_load_b128 runs.
#define BLKM 128
#define BLKN 64
#define BLKK 32
#define NKP  (BLKK/2)     // 16 packed K-pairs
#define LDP  (NKP + 4)    // 20 dwords/row = 80B: pads banks, keeps 16B align

__global__ __launch_bounds__(256)
void k_wmma_gemm(const unsigned short* __restrict__ A,
                 const unsigned short* __restrict__ B,
                 float* __restrict__ C,
                 int M, int N, int Kdim,
                 long long rsA, long long csA,
                 long long rsB, long long csB,
                 int ldc, float alpha, float betac)
{
  __shared__ __align__(16) unsigned int Apk[BLKM][LDP];
  __shared__ __align__(16) unsigned int Bpk[BLKN][LDP];

  const int mb = blockIdx.y * BLKM;
  const int nb = blockIdx.x * BLKN;
  const int tid  = threadIdx.x;
  const int lane = tid & 31;
  const int wave = tid >> 5;
  const int wm = wave >> 1;            // 0..3
  const int wn = wave & 1;             // 0..1
  const int half = lane >> 4;          // 0/1
  const int l16  = lane & 15;

  v8f acc[2][2] = {};

  for (int kb = 0; kb < Kdim; kb += BLKK) {
    // speculative prefetch of next K-slab (emits global_prefetch_b8)
    if (kb + BLKK < Kdim) {
      __builtin_prefetch((const void*)&A[(long long)(kb + BLKK + (tid >> 7)) * rsA +
                                         (long long)(mb + (tid & 127)) * csA], 0, 0);
      __builtin_prefetch((const void*)&B[(long long)(kb + BLKK + (tid >> 6)) * rsB +
                                         (long long)(nb + (tid & 63)) * csB], 0, 0);
    }

    // stage A: 128 x 16 packed dwords, 8 per thread (coalesced over m)
    #pragma unroll
    for (int e = 0; e < 8; ++e) {
      int widx = e * 256 + tid;
      int m  = widx & (BLKM - 1);
      int kp = widx >> 7;             // 0..15
      int gm = mb + m;
      int gk0 = kb + 2 * kp;
      unsigned int lo = 0, hi = 0;
      if (gm < M) {
        if (gk0 < Kdim)     lo = A[(long long)gk0 * rsA + (long long)gm * csA];
        if (gk0 + 1 < Kdim) hi = A[(long long)(gk0 + 1) * rsA + (long long)gm * csA];
      }
      Apk[m][kp] = lo | (hi << 16);
    }
    // stage B: 64 x 16 packed dwords, 4 per thread
    #pragma unroll
    for (int e = 0; e < 4; ++e) {
      int widx = e * 256 + tid;
      int n  = widx & (BLKN - 1);
      int kp = widx >> 6;             // 0..15
      int gn = nb + n;
      int gk0 = kb + 2 * kp;
      unsigned int lo = 0, hi = 0;
      if (gn < N) {
        if (gk0 < Kdim)     lo = B[(long long)gk0 * rsB + (long long)gn * csB];
        if (gk0 + 1 < Kdim) hi = B[(long long)(gk0 + 1) * rsB + (long long)gn * csB];
      }
      Bpk[n][kp] = lo | (hi << 16);
    }
    __syncthreads();

    // Fragments per ISA layout. A (16x32): VGPR j<4 -> kp=half*4+j,
    // j>=4 -> kp=8+half*4+(j-4)  => two contiguous 4-dword runs.
    // B (32x16): same as A: j<4 -> kp=half*4+j, j>=4 -> kp=8+half*4+(j-4).
    v16bf afrag[2], bfrag[2];
    #pragma unroll
    for (int s = 0; s < 2; ++s) {
      int m = wm * 32 + s * 16 + l16;
      v4i alo = *(const v4i*)&Apk[m][half * 4];
      v4i ahi = *(const v4i*)&Apk[m][8 + half * 4];
      afrag[s] = __builtin_bit_cast(
          v16bf, __builtin_shufflevector(alo, ahi, 0, 1, 2, 3, 4, 5, 6, 7));
      int n = wn * 32 + s * 16 + l16;
      v4i blo = *(const v4i*)&Bpk[n][half * 4];
      v4i bhi = *(const v4i*)&Bpk[n][8 + half * 4];
      bfrag[s] = __builtin_bit_cast(
          v16bf, __builtin_shufflevector(blo, bhi, 0, 1, 2, 3, 4, 5, 6, 7));
    }

    #pragma unroll
    for (int sa = 0; sa < 2; ++sa)
      #pragma unroll
      for (int sb = 0; sb < 2; ++sb)
        acc[sa][sb] = __builtin_amdgcn_wmma_f32_16x16x32_bf16(
            false, afrag[sa], false, bfrag[sb], (short)0, acc[sa][sb], false, false);

    __syncthreads();
  }

  // C layout: VGPR j -> row = j + 8*half; col = lane%16
  #pragma unroll
  for (int sa = 0; sa < 2; ++sa) {
    #pragma unroll
    for (int j = 0; j < 8; ++j) {
      int row = mb + wm * 32 + sa * 16 + j + half * 8;
      if (row >= M) continue;
      #pragma unroll
      for (int sb = 0; sb < 2; ++sb) {
        int col = nb + wn * 32 + sb * 16 + l16;
        if (col < N) {
          float v = alpha * acc[sa][sb][j];
          if (betac != 0.f) v += betac * C[(size_t)row * ldc + col];
          C[(size_t)row * ldc + col] = v;
        }
      }
    }
  }
}

// ---------------- feature extractor: data = relu(conv(x, fe_w) + fe_b) ----
__global__ void k_feature(const float* __restrict__ x, const float* __restrict__ fw,
                          const float* __restrict__ fb, float* __restrict__ data)
{
  int idx = blockIdx.x * blockDim.x + threadIdx.x;
  if (idx >= NBAT * INC * HH * WW) return;
  int w = idx % WW; int t = idx / WW;
  int h = t % HH;   t /= HH;
  int o = t % INC;  int n = t / INC;
  float acc = fb[o];
  for (int ci = 0; ci < 3; ++ci)
    for (int kh = 0; kh < 3; ++kh)
      for (int kw = 0; kw < 3; ++kw) {
        int ih = h + kh - 1, iw = w + kw - 1;
        if (ih < 0 || ih >= HH || iw < 0 || iw >= WW) continue;
        acc += x[((n * 3 + ci) * HH + ih) * WW + iw] *
               fw[((o * 3 + ci) * 3 + kh) * 3 + kw];
      }
  data[idx] = acc > 0.f ? acc : 0.f;
}

// ---------------- B tensor (bf16) for S = B^T B -------------------------
__global__ void k_buildB(const float* __restrict__ data, unsigned short* __restrict__ Bbf)
{
  int idx = blockIdx.x * blockDim.x + threadIdx.x;
  if (idx >= NROWB * SDIM) return;
  int col = idx % SDIM, row = idx / SDIM;
  int q = row % WO; int t = row / WO; int p = t % HO; int n = t / HO;
  int kw = col % KSZ; int kh = (col / KSZ) % KSZ; int i = col / (KSZ * KSZ);
  int hh = p - kh, w2 = q - kw;
  float v = 0.f;
  if (hh >= 0 && hh < HH && w2 >= 0 && w2 < WW)
    v = data[((n * INC + i) * HH + hh) * WW + w2];
  Bbf[idx] = f2bf(v);
}

__global__ void k_colsumB(const float* __restrict__ data, float* __restrict__ tB)
{
  int col = blockIdx.x * blockDim.x + threadIdx.x;
  if (col >= SDIM) return;
  int kw = col % KSZ; int kh = (col / KSZ) % KSZ; int i = col / (KSZ * KSZ);
  float acc = 0.f;
  for (int n = 0; n < NBAT; ++n)
    for (int p = 0; p < HO; ++p)
      for (int q = 0; q < WO; ++q) {
        int hh = p - kh, w2 = q - kw;
        if (hh >= 0 && hh < HH && w2 >= 0 && w2 < WW)
          acc += data[((n * INC + i) * HH + hh) * WW + w2];
      }
  tB[col] = acc;
}

// ---------------- f32 -> bf16 cast (grid-stride) ------------------------
__global__ void k_cast(const float* __restrict__ src, unsigned short* __restrict__ dst, size_t n)
{
  size_t i = (size_t)blockIdx.x * blockDim.x + threadIdx.x;
  size_t stride = (size_t)gridDim.x * blockDim.x;
  for (; i < n; i += stride) dst[i] = f2bf(src[i]);
}

// ---------------- T = beta^2 * Z^T Z (structured); also seeds G ---------
__global__ void k_TG(const float* __restrict__ S, const float* __restrict__ tB,
                     float* __restrict__ T, float* __restrict__ G)
{
  int idx = blockIdx.x * blockDim.x + threadIdx.x;
  if (idx >= NW * NW) return;
  int c = idx % NW, r = idx / NW;
  const float beta2 = 1.f / (float)NW;
  float v = 0.f;
  if (r < NWC && c < NWC) {
    int rr = r % (OUTC * 9); int ir = r / (OUTC * 9); int dr = rr / 9; int klr = rr % 9;
    int cc = c % (OUTC * 9); int ic = c / (OUTC * 9); int dc = cc / 9; int klc = cc % 9;
    if (dr == dc) v = S[(ir * 9 + klr) * SDIM + (ic * 9 + klc)];
  } else if (r < NWC) {
    int rr = r % (OUTC * 9); int ir = r / (OUTC * 9); int dr = rr / 9; int klr = rr % 9;
    if (dr == c - NWC) v = tB[ir * 9 + klr];
  } else if (c < NWC) {
    int cc = c % (OUTC * 9); int ic = c / (OUTC * 9); int dc = cc / 9; int klc = cc % 9;
    if (dc == r - NWC) v = tB[ic * 9 + klc];
  } else if (r == c) {
    v = (float)NROWB;
  }
  v *= beta2;
  T[idx] = v;
  G[idx] = v;
}

__global__ void k_setI(float* __restrict__ V)
{
  int idx = blockIdx.x * blockDim.x + threadIdx.x;
  if (idx >= NW * NW) return;
  V[idx] = ((idx / NW) == (idx % NW)) ? 1.f : 0.f;
}

// ---------------- parallel-ordered cyclic Jacobi eigensolver ------------
__global__ void k_jangles(const float* __restrict__ G, float* __restrict__ jc,
                          float* __restrict__ js, int* __restrict__ jp,
                          int* __restrict__ jq, int round)
{
  int i = blockIdx.x * blockDim.x + threadIdx.x;
  if (i >= NPAIRS) return;
  const int m = NW - 1;
  int p = (i == 0) ? 0 : ((i - 1 + round) % m) + 1;
  int sl = NW - 1 - i;
  int q = ((sl - 1 + round) % m) + 1;
  if (p > q) { int t = p; p = q; q = t; }
  float app = G[(size_t)p * NW + p];
  float aqq = G[(size_t)q * NW + q];
  float apq = G[(size_t)p * NW + q];
  float c = 1.f, s = 0.f;
  if (fabsf(apq) > 1e-30f) {
    float tau = (aqq - app) / (2.f * apq);
    float tt = ((tau >= 0.f) ? 1.f : -1.f) / (fabsf(tau) + sqrtf(1.f + tau * tau));
    c = rsqrtf(1.f + tt * tt);
    s = tt * c;
  }
  jc[i] = c; js[i] = s; jp[i] = p; jq[i] = q;
}

__global__ void k_jrows(float* __restrict__ G, const float* __restrict__ jc,
                        const float* __restrict__ js, const int* __restrict__ jp,
                        const int* __restrict__ jq)
{
  int idx = blockIdx.x * blockDim.x + threadIdx.x;
  if (idx >= NPAIRS * NW) return;
  int col = idx % NW, pr = idx / NW;
  int p = jp[pr], q = jq[pr];
  float c = jc[pr], s = js[pr];
  float gp = G[(size_t)p * NW + col];
  float gq = G[(size_t)q * NW + col];
  G[(size_t)p * NW + col] = c * gp - s * gq;
  G[(size_t)q * NW + col] = s * gp + c * gq;
}

__global__ void k_jcols(float* __restrict__ G, float* __restrict__ V,
                        const float* __restrict__ jc, const float* __restrict__ js,
                        const int* __restrict__ jp, const int* __restrict__ jq)
{
  int idx = blockIdx.x * blockDim.x + threadIdx.x;
  if (idx >= NPAIRS * NW) return;
  int row = idx % NW, pr = idx / NW;
  int p = jp[pr], q = jq[pr];
  float c = jc[pr], s = js[pr];
  float gp = G[(size_t)row * NW + p];
  float gq = G[(size_t)row * NW + q];
  G[(size_t)row * NW + p] = c * gp - s * gq;
  G[(size_t)row * NW + q] = s * gp + c * gq;
  float vp = V[(size_t)row * NW + p];
  float vq = V[(size_t)row * NW + q];
  V[(size_t)row * NW + p] = c * vp - s * vq;
  V[(size_t)row * NW + q] = s * vp + c * vq;
}

__global__ void k_diag(const float* __restrict__ G, float* __restrict__ s2)
{
  int i = blockIdx.x * blockDim.x + threadIdx.x;
  if (i < NW) s2[i] = G[(size_t)i * NW + i];
}

// ---------------- structured Z products ---------------------------------
__global__ void k_zw(const float* __restrict__ data, const float* __restrict__ Wv,
                     const float* __restrict__ cin, float* __restrict__ Zwc)
{
  int row = blockIdx.x * blockDim.x + threadIdx.x;
  if (row >= NUMZ) return;
  int q = row % WO; int t = row / WO; int p = t % HO; t /= HO;
  int ch = t % OUTC; int n = t / OUTC;
  const float beta_ = 1.f / sqrtf((float)NW);
  float acc = Wv[NWC + ch];
  for (int i = 0; i < INC; ++i)
    for (int kh = 0; kh < KSZ; ++kh)
      for (int kw = 0; kw < KSZ; ++kw) {
        int hh = p - kh, w2 = q - kw;
        if (hh < 0 || hh >= HH || w2 < 0 || w2 >= WW) continue;
        acc += data[((n * INC + i) * HH + hh) * WW + w2] *
               Wv[((i * OUTC + ch) * KSZ + kh) * KSZ + kw];
      }
  Zwc[row] = beta_ * (acc - cin[row]);
}

__global__ void k_rvec(const float* __restrict__ data, const float* __restrict__ Zwc,
                       float* __restrict__ rv)
{
  int col = blockIdx.x * blockDim.x + threadIdx.x;
  if (col >= NW) return;
  const float beta_ = 1.f / sqrtf((float)NW);
  float acc = 0.f;
  if (col < NWC) {
    int cc = col % (OUTC * 9); int i = col / (OUTC * 9);
    int d = cc / 9; int kl = cc % 9; int kh = kl / 3, kw = kl % 3;
    for (int n = 0; n < NBAT; ++n)
      for (int p = 0; p < HO; ++p)
        for (int q = 0; q < WO; ++q) {
          int hh = p - kh, w2 = q - kw;
          if (hh < 0 || hh >= HH || w2 < 0 || w2 >= WW) continue;
          acc += data[((n * INC + i) * HH + hh) * WW + w2] *
                 Zwc[((n * OUTC + d) * HO + p) * WO + q];
        }
  } else {
    int d = col - NWC;
    for (int n = 0; n < NBAT; ++n)
      for (int pq = 0; pq < HO * WO; ++pq)
        acc += Zwc[(n * OUTC + d) * HO * WO + pq];
  }
  rv[col] = beta_ * acc;
}

__global__ void k_zero(float* __restrict__ scal)
{
  int i = threadIdx.x;
  if (i < 64) scal[i] = 0.f;
}

__global__ void k_rho(const float* __restrict__ Zwc, float* __restrict__ scal)
{
  __shared__ float red[256];
  int idx = blockIdx.x * 256 + threadIdx.x;
  float v = 0.f;
  if (idx < NUMZ) { float z = Zwc[idx]; v = z * z; }
  red[threadIdx.x] = v; __syncthreads();
  for (int o = 128; o > 0; o >>= 1) {
    if (threadIdx.x < o) red[threadIdx.x] += red[threadIdx.x + o];
    __syncthreads();
  }
  if (threadIdx.x == 0) atomicAdd(&scal[0], red[0]);
}

__global__ void k_gemv(const float* __restrict__ V, const float* __restrict__ vec,
                       float* __restrict__ outv)
{
  int j = blockIdx.x * blockDim.x + threadIdx.x;
  if (j >= NW) return;
  float acc = 0.f;
  for (int i = 0; i < NW; ++i) acc += V[(size_t)i * NW + j] * vec[i];
  outv[j] = acc;
}

__global__ void k_coeff(const float* __restrict__ s2, const float* __restrict__ VTr,
                        const float* __restrict__ VTw, const float* __restrict__ Hm,
                        float* __restrict__ coeff, float* __restrict__ scal)
{
  int idx = blockIdx.x * blockDim.x + threadIdx.x;
  if (idx >= NLAM * NW) return;
  int j = idx % NW; int l = idx / NW;
  float lam = lamval(l);
  float d = 1.f / (s2[j] + SUM_LAMBDA_C + lam);
  coeff[idx] = d * (VTr[j] + lam * VTw[j]);
  atomicAdd(&scal[1 + l], d * Hm[(size_t)j * NW + j]);   // trace term (coln2 = diag H)
}

__global__ void k_hcoeff(const float* __restrict__ Hm, const float* __restrict__ coeff,
                         float* __restrict__ Hc)
{
  int idx = blockIdx.x * blockDim.x + threadIdx.x;
  if (idx >= NLAM * NW) return;
  int i = idx % NW; int l = idx / NW;
  const float* cf = coeff + (size_t)l * NW;
  const float* hrow = Hm + (size_t)i * NW;
  float acc = 0.f;
  for (int j = 0; j < NW; ++j) acc += hrow[j] * cf[j];
  Hc[idx] = acc;
}

__global__ void k_gcv(const float* __restrict__ coeff, const float* __restrict__ Hc,
                      const float* __restrict__ VTr, float* __restrict__ scal,
                      int* __restrict__ bestI)
{
  __shared__ float red[256];
  __shared__ float gv[NLAM];
  int tid = threadIdx.x;
  for (int l = 0; l < NLAM; ++l) {
    float lin = 0.f, quad = 0.f;
    for (int j = tid; j < NW; j += 256) {
      float cl = coeff[l * NW + j];
      lin  += cl * VTr[j];
      quad += cl * Hc[l * NW + j];
    }
    red[tid] = lin; __syncthreads();
    for (int o = 128; o > 0; o >>= 1) { if (tid < o) red[tid] += red[tid + o]; __syncthreads(); }
    float lint = red[0]; __syncthreads();
    red[tid] = quad; __syncthreads();
    for (int o = 128; o > 0; o >>= 1) { if (tid < o) red[tid] += red[tid + o]; __syncthreads(); }
    if (tid == 0) {
      float quadt = red[0];
      float resn = scal[0] - 2.f * lint + quadt;     // ||Zw_c||^2 - 2 c^T VTr + c^T H c
      float tl = scal[1 + l];
      float nz = (float)NUMZ;
      float den = nz - tl;
      gv[l] = nz * resn / (den * den);
    }
    __syncthreads();
  }
  if (tid == 0) {
    int best = 0; float bg = gv[0];
    for (int l = 1; l < NLAM; ++l) if (gv[l] < bg) { bg = gv[l]; best = l; }
    bestI[0] = best;
    scal[12] = lamval(best);
  }
}

__global__ void k_wnew(const float* __restrict__ V, const float* __restrict__ coeff,
                       const float* __restrict__ Win, const int* __restrict__ bestI,
                       float* __restrict__ wnew)
{
  int i = blockIdx.x * blockDim.x + threadIdx.x;
  if (i >= NW) return;
  const float* cf = coeff + (size_t)bestI[0] * NW;
  float acc = 0.f;
  for (int j = 0; j < NW; ++j) acc += V[(size_t)i * NW + j] * cf[j];
  wnew[i] = Win[i] - acc;
}

// ---------------- final output: out = Z @ w_new (== conv_transpose) -----
__global__ void k_out(const float* __restrict__ data, const float* __restrict__ wnew,
                      float* __restrict__ out)
{
  int row = blockIdx.x * blockDim.x + threadIdx.x;
  if (row >= NUMZ) return;
  int q = row % WO; int t = row / WO; int p = t % HO; t /= HO;
  int ch = t % OUTC; int n = t / OUTC;
  float acc = wnew[NWC + ch];
  for (int i = 0; i < INC; ++i)
    for (int kh = 0; kh < KSZ; ++kh)
      for (int kw = 0; kw < KSZ; ++kw) {
        int hh = p - kh, w2 = q - kw;
        if (hh < 0 || hh >= HH || w2 < 0 || w2 >= WW) continue;
        acc += data[((n * INC + i) * HH + hh) * WW + w2] *
               wnew[((i * OUTC + ch) * KSZ + kh) * KSZ + kw];
      }
  out[row] = acc;
}

// ---------------- host orchestration ------------------------------------
extern "C" void kernel_launch(void* const* d_in, const int* in_sizes, int n_in,
                              void* d_out, int out_size, void* d_ws, size_t ws_size,
                              hipStream_t stream)
{
  const float* x     = (const float*)d_in[0];
  const float* cin   = (const float*)d_in[1];
  const float* Win   = (const float*)d_in[2];
  const float* Mprev = (const float*)d_in[3];
  const float* few   = (const float*)d_in[4];
  const float* feb   = (const float*)d_in[5];
  float* out = (float*)d_out;
  (void)in_sizes; (void)n_in; (void)out_size; (void)ws_size;

  char* ws = (char*)d_ws;
  size_t off = 0;
  auto alloc = [&](size_t bytes) -> void* {
    off = (off + 255) & ~(size_t)255;
    void* p = (void*)(ws + off);
    off += bytes;
    return p;
  };

  float*          data = (float*)alloc((size_t)NBAT * INC * HH * WW * 4);
  unsigned short* Bbf  = (unsigned short*)alloc((size_t)NROWB * SDIM * 2);
  float*          tB   = (float*)alloc((size_t)SDIM * 4);
  unsigned short* Abf  = (unsigned short*)alloc((size_t)NUMZ * NW * 2);
  float*          S    = (float*)alloc((size_t)SDIM * SDIM * 4);
  float*          T    = (float*)alloc((size_t)NW * NW * 4);
  float*          G    = (float*)alloc((size_t)NW * NW * 4);
  float*          V    = (float*)alloc((size_t)NW * NW * 4);
  float*          P    = (float*)alloc((size_t)NW * NW * 4);
  float*          Hm   = (float*)alloc((size_t)NW * NW * 4);
  unsigned short* Vbf  = (unsigned short*)alloc((size_t)NW * NW * 2);
  unsigned short* Tbf  = (unsigned short*)alloc((size_t)NW * NW * 2);
  unsigned short* Pbf  = (unsigned short*)alloc((size_t)NW * NW * 2);
  float*          jc   = (float*)alloc((size_t)NPAIRS * 4);
  float*          js   = (float*)alloc((size_t)NPAIRS * 4);
  int*            jp   = (int*)alloc((size_t)NPAIRS * 4);
  int*            jq   = (int*)alloc((size_t)NPAIRS * 4);
  float*          s2   = (float*)alloc((size_t)NW * 4);
  float*          Zwc  = (float*)alloc((size_t)NUMZ * 4);
  float*          rv   = (float*)alloc((size_t)NW * 4);
  float*          VTr  = (float*)alloc((size_t)NW * 4);
  float*          VTw  = (float*)alloc((size_t)NW * 4);
  float*          coeff= (float*)alloc((size_t)NLAM * NW * 4);
  float*          Hc   = (float*)alloc((size_t)NLAM * NW * 4);
  float*          wnew = (float*)alloc((size_t)NW * 4);
  float*          scal = (float*)alloc(64 * 4);
  int*            bestI= (int*)(scal + 48);

  const float beta2 = 1.f / (float)NW;
  const float* Arows = Mprev + (size_t)NUMZ * NW;     // M_prev[NUMZ:]

  // 1. feature extractor
  k_feature<<<(NBAT * INC * HH * WW + 255) / 256, 256, 0, stream>>>(x, few, feb, data);

  // 2. structured-Z precursors
  k_buildB<<<(NROWB * SDIM + 255) / 256, 256, 0, stream>>>(data, Bbf);
  k_colsumB<<<1, 256, 0, stream>>>(data, tB);
  k_cast<<<4096, 256, 0, stream>>>(Arows, Abf, (size_t)NUMZ * NW);

  // 3. S = B^T B  (WMMA, 144x144, K=2592)
  {
    dim3 g((SDIM + BLKN - 1) / BLKN, (SDIM + BLKM - 1) / BLKM);
    k_wmma_gemm<<<g, 256, 0, stream>>>(Bbf, Bbf, S, SDIM, SDIM, NROWB,
                                       SDIM, 1, SDIM, 1, SDIM, 1.f, 0.f);
  }

  // 4. T = beta^2 Z^T Z (structured), seed G = T
  k_TG<<<(NW * NW + 255) / 256, 256, 0, stream>>>(S, tB, T, G);

  // 5. G += beta^2 * A^T A   (the dominant 0.45-TFLOP WMMA GEMM)
  dim3 gNW((NW + BLKN - 1) / BLKN, (NW + BLKM - 1) / BLKM);
  k_wmma_gemm<<<gNW, 256, 0, stream>>>(Abf, Abf, G, NW, NW, NUMZ,
                                       NW, 1, NW, 1, NW, beta2, 1.f);

  // 6. eigh(G) via parallel-ordered cyclic Jacobi; V accumulates eigenvectors
  k_setI<<<(NW * NW + 255) / 256, 256, 0, stream>>>(V);
  for (int sweep = 0; sweep < NSWEEPS; ++sweep) {
    for (int r = 0; r < NW - 1; ++r) {
      k_jangles<<<(NPAIRS + 255) / 256, 256, 0, stream>>>(G, jc, js, jp, jq, r);
      k_jrows<<<(NPAIRS * NW + 255) / 256, 256, 0, stream>>>(G, jc, js, jp, jq);
      k_jcols<<<(NPAIRS * NW + 255) / 256, 256, 0, stream>>>(G, V, jc, js, jp, jq);
    }
  }
  k_diag<<<(NW + 255) / 256, 256, 0, stream>>>(G, s2);

  // 7. H = V^T (Zs^T Zs) V = V^T T V  (two WMMA GEMMs; diag(H) = coln2)
  k_cast<<<1024, 256, 0, stream>>>(V, Vbf, (size_t)NW * NW);
  k_cast<<<1024, 256, 0, stream>>>(T, Tbf, (size_t)NW * NW);
  k_wmma_gemm<<<gNW, 256, 0, stream>>>(Vbf, Tbf, P, NW, NW, NW,
                                       NW, 1, NW, 1, NW, 1.f, 0.f);      // P = V^T T
  k_cast<<<1024, 256, 0, stream>>>(P, Pbf, (size_t)NW * NW);
  k_wmma_gemm<<<gNW, 256, 0, stream>>>(Pbf, Vbf, Hm, NW, NW, NW,
                                       1, NW, NW, 1, NW, 1.f, 0.f);      // H = P V

  // 8. small dense algebra for the GCV ridge solve
  k_zero<<<1, 64, 0, stream>>>(scal);
  k_zw<<<(NUMZ + 255) / 256, 256, 0, stream>>>(data, Win, cin, Zwc);
  k_rvec<<<(NW + 255) / 256, 256, 0, stream>>>(data, Zwc, rv);
  k_rho<<<(NUMZ + 255) / 256, 256, 0, stream>>>(Zwc, scal);
  k_gemv<<<(NW + 255) / 256, 256, 0, stream>>>(V, rv, VTr);
  k_gemv<<<(NW + 255) / 256, 256, 0, stream>>>(V, Win, VTw);
  k_coeff<<<(NLAM * NW + 255) / 256, 256, 0, stream>>>(s2, VTr, VTw, Hm, coeff, scal);
  k_hcoeff<<<(NLAM * NW + 255) / 256, 256, 0, stream>>>(Hm, coeff, Hc);
  k_gcv<<<1, 256, 0, stream>>>(coeff, Hc, VTr, scal, bestI);
  k_wnew<<<(NW + 255) / 256, 256, 0, stream>>>(V, coeff, Win, bestI, wnew);

  // 9. output = conv_transpose_apply(data, w_new) == Z @ w_new
  k_out<<<(NUMZ + 255) / 256, 256, 0, stream>>>(data, wnew, out);
}
